// RNNModule_64106681860256
// MI455X (gfx1250) — matrix operand order repaired and checked by reading, not compiled, not run on hardware
//
#include <hip/hip_runtime.h>
#include <hip/hip_bf16.h>

// ---------------------------------------------------------------------------
// Leaky tanh RNN on MI455X (gfx1250), persistent cooperative kernel.
//   x_{t+1} = x_t + a*(-x_t + r_t @ Wm^T + u_t + bias),  r = tanh(x)
// GEMM per step: C[b,j] = sum_i r[b,i] * Wm[j,i], B=64, N=K=1024.
// 64 WGs x 4 wave32: WG owns 16 output columns (j-tile), wave owns 16 batch
// rows. bf16 WMMA 16x16x32 with f32 accumulation; masked W slice staged to
// LDS once in B-fragment layout; r ping-pongs through L2 as bf16.
// x state lives entirely in VGPRs (each (b,j) owned by one lane for all T).
// ---------------------------------------------------------------------------

#define T_STEPS   512
#define BB        64
#define NN        1024
#define BN        (BB * NN)
#define ALPHA_F   0.2f
#define NBLOCKS   64        // N / 16 column tiles
#define NTHREADS  128       // 4 wave32

typedef __attribute__((ext_vector_type(16))) __bf16 v16bf;
typedef __attribute__((ext_vector_type(8)))  float  v8f;

union Frag { uint4 u[2]; v16bf v; };

__device__ inline unsigned short f2bf(float x) {
  __hip_bfloat16 h = __float2bfloat16(x);   // round-to-nearest-even
  return __builtin_bit_cast(unsigned short, h);
}

// Branchless tanh: 1 - 2/(e^{2x}+1) via native v_exp_f32 / v_rcp_f32.
// Saturates correctly to +/-1 for large |x|; ~1e-6 rel error (far below the
// ~1e-3 noise injected by bf16 GEMM operands).
__device__ inline float tanh_fast(float x) {
  const float e = __expf(2.0f * x);
  return 1.0f - 2.0f * __builtin_amdgcn_rcpf(e + 1.0f);
}

// ---- prep: reset barrier; rates[0] = tanh(init_x); bf16 ping buffer --------
__global__ __launch_bounds__(256) void rnn_prep(
    const float* __restrict__ init_x, unsigned short* __restrict__ rbuf0,
    float* __restrict__ out, unsigned* __restrict__ cnt) {
  const int i = blockIdx.x * blockDim.x + threadIdx.x;
  if (i == 0) *cnt = 0u;                 // reset grid barrier every launch
  if (i < BN) {
    const float rv = tanhf(init_x[i]);
    out[i]   = rv;                       // rates[0]
    rbuf0[i] = f2bf(rv);
  }
}

// ---- persistent step kernel ------------------------------------------------
__global__ __launch_bounds__(NTHREADS) void rnn_persistent(
    const float* __restrict__ inputs,   // [T,B,N]
    const float* __restrict__ Wrec,     // [N,N] row-major [j][i]
    const float* __restrict__ bias,     // [N]
    const unsigned char* __restrict__ mask, // [N,N] bool bytes
    const float* __restrict__ init_x,   // [B,N]
    unsigned short* __restrict__ rbuf0, // [B,N] bf16 ping
    unsigned short* __restrict__ rbuf1, // [B,N] bf16 pong
    float* __restrict__ out,            // [T+1,B,N]
    unsigned* __restrict__ cnt)         // grid barrier counter
{
  // B-operand fragments: chunk c (K=32c..32c+31), lane L holds column
  // n=L%16, K = 32c + (L/16)*16 + 0..15 (16 consecutive bf16 = 32B).
  __shared__ __align__(16) unsigned short lds_w[32 * 32 * 16]; // 32 KB

  const int tid  = threadIdx.x;
  const int j0   = blockIdx.x * 16;   // column tile
  const int wave = tid >> 5;          // batch tile 0..3
  const int lane = tid & 31;
  const int half = lane >> 4;
  const int nsub = lane & 15;
  const int b0   = wave * 16;

  // One-time: masked W slice -> bf16 -> LDS in fragment order.
  for (int p = tid; p < 32 * 32; p += NTHREADS) {
    const int c = p >> 5, L = p & 31;
    const int j  = j0 + (L & 15);
    const int ib = (c << 5) + ((L >> 4) << 4);
    const float*         wr = Wrec + (size_t)j * NN + ib;
    const unsigned char* mr = mask + (size_t)j * NN + ib;
    unsigned short* dst = lds_w + p * 16;
#pragma unroll
    for (int q = 0; q < 16; ++q)
      dst[q] = f2bf(mr[q] ? wr[q] : 0.0f);
  }
  __syncthreads();

  // This lane permanently owns elements (b,j) with b = b0+half*8+g, j fixed.
  const int    j    = j0 + nsub;
  const float  bj   = bias[j];
  const size_t idx0 = (size_t)(b0 + half * 8) * NN + j;

  float xreg[8];                         // x state lives in VGPRs for all T
#pragma unroll
  for (int g = 0; g < 8; ++g) xreg[g] = init_x[idx0 + (size_t)g * NN];

  for (int t = 0; t < T_STEPS; ++t) {
    const unsigned short* rp = (t & 1) ? rbuf1 : rbuf0;
    unsigned short*       rn = (t & 1) ? rbuf0 : rbuf1;

    // Hoist u_t loads (only HBM read stream) so the GEMM hides their latency.
    const size_t ubase = (size_t)t * BN;
    float uv[8];
    const float* up = inputs + ubase + idx0;
#pragma unroll
    for (int g = 0; g < 8; ++g) uv[g] = up[(size_t)g * NN];

    // A fragment base: lane holds row b0+(lane&15); contiguous K order
    // matching B: v0-7 = K + half*16 + 0..15.
    const unsigned short* arow = rp + (size_t)(b0 + nsub) * NN + half * 16;

    v8f acc = {0.f, 0.f, 0.f, 0.f, 0.f, 0.f, 0.f, 0.f};
#pragma unroll 8
    for (int c = 0; c < 32; ++c) {
      Frag a, b;
      const unsigned short* ap = arow + (c << 5);
      a.u[0] = *reinterpret_cast<const uint4*>(ap);
      a.u[1] = *reinterpret_cast<const uint4*>(ap + 8);
      const unsigned short* bp = lds_w + (((c << 5) + lane) << 4);
      b.u[0] = *reinterpret_cast<const uint4*>(bp);
      b.u[1] = *reinterpret_cast<const uint4*>(bp + 8);
      acc = __builtin_amdgcn_wmma_f32_16x16x32_bf16(
          false, a.v, false, b.v, (short)0, acc, false, false);
    }

    // Epilogue: C/D layout -> lane owns rows m = half*8 + g, col n = lane&15.
#pragma unroll
    for (int g = 0; g < 8; ++g) {
      const size_t idx = idx0 + (size_t)g * NN;
      const float  xn  = xreg[g] + ALPHA_F * (acc[g] + uv[g] + bj - xreg[g]);
      xreg[g] = xn;
      const float rv = tanh_fast(xn);
      out[ubase + BN + idx] = rv;    // rates[t+1]
      rn[idx] = f2bf(rv);
    }

    // Device-wide barrier: make r_t visible before step t+1 reads it.
    __threadfence();
    __syncthreads();
    if (tid == 0) {
      __hip_atomic_fetch_add(cnt, 1u, __ATOMIC_ACQ_REL,
                             __HIP_MEMORY_SCOPE_AGENT);
      const unsigned want = (unsigned)NBLOCKS * (unsigned)(t + 1);
      while (__hip_atomic_load(cnt, __ATOMIC_ACQUIRE,
                               __HIP_MEMORY_SCOPE_AGENT) < want)
        __builtin_amdgcn_s_sleep(1);
    }
    __syncthreads();
    __threadfence();
  }
}

// ---------------------------------------------------------------------------
extern "C" void kernel_launch(void* const* d_in, const int* in_sizes, int n_in,
                              void* d_out, int out_size, void* d_ws,
                              size_t ws_size, hipStream_t stream) {
  const float*         inputs = (const float*)d_in[0];         // [512,64,1024]
  const float*         Wrec   = (const float*)d_in[1];         // [1024,1024]
  const float*         bias   = (const float*)d_in[2];         // [1,1024]
  const float*         init_x = (const float*)d_in[3];         // [64,1024]
  const unsigned char* mask   = (const unsigned char*)d_in[4]; // bool bytes
  float* out = (float*)d_out;                                  // [513,64,1024]

  char* ws = (char*)d_ws;
  unsigned short* rbuf0 = (unsigned short*)ws;                 // 128 KB
  unsigned short* rbuf1 = (unsigned short*)(ws + 131072);      // 128 KB
  unsigned*       cnt   = (unsigned*)(ws + 262144);

  rnn_prep<<<BN / 256, 256, 0, stream>>>(init_x, rbuf0, out, cnt);
  rnn_persistent<<<NBLOCKS, NTHREADS, 0, stream>>>(
      inputs, Wrec, bias, mask, init_x, rbuf0, rbuf1, out, cnt);
}
